// BiLSTM_ChainCRF_35905926594954
// MI455X (gfx1250) — compile-verified, matrix operand order checked by reading, not run on hardware
//
#include <hip/hip_runtime.h>
#include <hip/hip_bf16.h>
#include <math.h>

// ---------------- problem constants (match reference) ----------------
#define SEQ   4096
#define EMB   1024
#define HID   1024
#define Hh    512          // H = HID/2
#define G4    2048         // 4*H
#define NTAG  34
#define START 32
#define STOP  33
#define NEGV  (-10000.0f)

// LDS slab geometry for the TDM-fed GEMM: rows of 32 bf16 (64B) padded by the
// TDM to an 80B pitch (pad_interval = 16 dwords, pad_amount = 4 dwords) so the
// 16-lane fragment reads hit 64 distinct banks instead of 4-way conflicting.
#define ROWPITCH 80
#define A_SLAB   (64  * ROWPITCH)   // 64 rows  x 32 bf16
#define B_SLAB   (128 * ROWPITCH)   // 128 rows x 32 bf16

// ---------------- WMMA fragment / TDM descriptor types ----------------
typedef __bf16 v16bf __attribute__((ext_vector_type(16)));
typedef float  v8f   __attribute__((ext_vector_type(8)));
typedef unsigned int u32x4 __attribute__((ext_vector_type(4)));
typedef int          i32x4 __attribute__((ext_vector_type(4)));
typedef int          i32x8 __attribute__((ext_vector_type(8)));

union Frag { v16bf v; uint4 q[2]; };

// ---- D# group 0: count=1, lds byte addr, 57-bit global addr, type=2 ----
__device__ __forceinline__ u32x4 tdm_g0(unsigned lds_off, unsigned long long ga) {
    u32x4 g;
    g.x = 1u;                                             // count = 1 (valid)
    g.y = lds_off;                                        // lds_addr[31:0]
    g.z = (unsigned)ga;                                   // global_addr[31:0]
    g.w = (unsigned)((ga >> 32) & 0x1FFFFFFu)             // global_addr[56:32]
        | 0x80000000u;                                    // type = 2 ("image")
    return g;
}

// ---- D# group 1: 2B elements, 2D tile, dim0 stride, LDS pad 64B->80B ----
__device__ __forceinline__ i32x8 tdm_g1(unsigned td0, unsigned td1,
                                        unsigned tile0, unsigned tile1,
                                        unsigned stride) {
    i32x8 g;
    g[0] = (int)(0x00010000u          // data_size = 1 -> 2 bytes
               | (1u << 20)           // pad_enable
               | (3u << 22)           // pad_interval = 16 dwords (64B)
               | (3u << 25));         // pad_amount   = 4 dwords  (16B)
    g[1] = (int)((td0 & 0xFFFFu) << 16);                          // tensor_dim0 lo
    g[2] = (int)(((td0 >> 16) & 0xFFFFu) | ((td1 & 0xFFFFu) << 16)); // td0 hi | td1 lo
    g[3] = (int)(((td1 >> 16) & 0xFFFFu) | (tile0 << 16));        // td1 hi | tile_dim0
    g[4] = (int)(tile1 & 0xFFFFu);                                // tile_dim1 (tile_dim2=0)
    g[5] = (int)stride;                                           // tensor_dim0_stride lo32
    g[6] = 0;                                                     // stride hi | dim1_stride lo
    g[7] = 0;
    return g;
}

__device__ __forceinline__ void tdm_load(u32x4 g0, i32x8 g1) {
    i32x4 z4 = {0, 0, 0, 0};
#if defined(__clang_major__) && (__clang_major__ >= 23)
    i32x8 z8 = {0, 0, 0, 0, 0, 0, 0, 0};
    __builtin_amdgcn_tensor_load_to_lds(g0, g1, z4, z4, z8, 0);
#else
    __builtin_amdgcn_tensor_load_to_lds(g0, g1, z4, z4, 0);
#endif
}

// ---------------- f32 -> bf16 conversion ----------------
__global__ void f32_to_bf16_kernel(const float* __restrict__ src,
                                   __bf16* __restrict__ dst, int n) {
    int i = blockIdx.x * blockDim.x + threadIdx.x;
    if (i < n) dst[i] = (__bf16)src[i];
}

// ---------------- embedding gather (f32 table -> bf16 rows) ----------------
__global__ void embed_gather_kernel(const int* __restrict__ sent,
                                    const float* __restrict__ embed,
                                    __bf16* __restrict__ x) {
    int t = blockIdx.x;
    int tok = sent[t];
    const float* src = embed + (size_t)tok * EMB;
    __bf16* dst = x + (size_t)t * EMB;
    for (int i = threadIdx.x; i < EMB; i += blockDim.x)
        dst[i] = (__bf16)src[i];
}

// ---------------- gate GEMM: xg = x @ W_ih^T + (b_ih + b_hh) ----------------
// TDM double-buffered pipeline: wave 0 DMAs the 64x32 A slab and 128x32 B slab
// for K-step k+1 into LDS (TENSORcnt) while all 8 waves run WMMA on K-step k
// from LDS. Each wave owns a 32x32 C tile (2x2 of 16x16), 128 wmma per wave.
__global__ __launch_bounds__(256)
void gate_gemm_kernel(const __bf16* __restrict__ x,       // [SEQ][EMB]
                      const __bf16* __restrict__ wih_f,   // [G4][EMB]
                      const __bf16* __restrict__ wih_b,   // [G4][EMB]
                      const float* __restrict__ bih_f, const float* __restrict__ bhh_f,
                      const float* __restrict__ bih_b, const float* __restrict__ bhh_b,
                      float* __restrict__ xg_f, float* __restrict__ xg_b) {
    __shared__ __align__(16) char smem[2 * (A_SLAB + B_SLAB)];

    const int dir = blockIdx.z;
    const __bf16* W  = dir ? wih_b : wih_f;
    const float*  bi = dir ? bih_b : bih_f;
    const float*  bh = dir ? bhh_b : bhh_f;
    float*        xg = dir ? xg_b  : xg_f;

    const int lane = threadIdx.x & 31;
    const int wid  = threadIdx.x >> 5;
    const int half = lane >> 4;
    const int r    = lane & 15;

    const int m_blk  = blockIdx.x * 64;
    const int n_blk  = blockIdx.y * 128;
    const int m_base = m_blk + (wid >> 2) * 32;   // 2 wave-rows
    const int n_base = n_blk + (wid & 3) * 32;    // 4 wave-cols

    char* bufA[2] = { smem,                     smem + (A_SLAB + B_SLAB) };
    char* bufB[2] = { smem + A_SLAB,            smem + (A_SLAB + B_SLAB) + A_SLAB };

    // accumulators initialized with broadcast bias (all 8 rows share column N)
    v8f acc[2][2];
    for (int in = 0; in < 2; ++in) {
        float b = bi[n_base + in * 16 + r] + bh[n_base + in * 16 + r];
        v8f v;
        for (int e = 0; e < 8; ++e) v[e] = b;
        acc[0][in] = v;
        acc[1][in] = v;
    }

    // TDM tile descriptors (group 1 is K-step invariant)
    const i32x8 g1A = tdm_g1(EMB, SEQ, 32, 64,  EMB);
    const i32x8 g1B = tdm_g1(EMB, G4,  32, 128, EMB);
    const unsigned long long gaA0 = (unsigned long long)(uintptr_t)(x + (size_t)m_blk * EMB);
    const unsigned long long gaB0 = (unsigned long long)(uintptr_t)(W + (size_t)n_blk * EMB);

    // prologue: DMA K-step 0 into buffer 0
    if (wid == 0) {
        tdm_load(tdm_g0((unsigned)(uintptr_t)bufA[0], gaA0), g1A);
        tdm_load(tdm_g0((unsigned)(uintptr_t)bufB[0], gaB0), g1B);
    }

    const int NSTEP = EMB / 32;
    for (int s = 0; s < NSTEP; ++s) {
        const int b = s & 1;
        if (wid == 0) {
            if (s + 1 < NSTEP) {
                // DMA next K slab into the other buffer, then wait for this one
                unsigned long long koff = (unsigned long long)(s + 1) * 32 * 2;
                tdm_load(tdm_g0((unsigned)(uintptr_t)bufA[b ^ 1], gaA0 + koff), g1A);
                tdm_load(tdm_g0((unsigned)(uintptr_t)bufB[b ^ 1], gaB0 + koff), g1B);
                __builtin_amdgcn_s_wait_tensorcnt(2);
            } else {
                __builtin_amdgcn_s_wait_tensorcnt(0);
            }
        }
        __syncthreads();   // LDS buffer b is ready for all waves

        // fragment reads from LDS (ds_load_b128, bank-conflict-free 80B pitch)
        Frag a[2], bb[2];
        const char* sA = bufA[b];
        const char* sB = bufB[b];
        const int arow0 = (wid >> 2) * 32 + r;
        const int brow0 = (wid & 3) * 32 + r;
        for (int im = 0; im < 2; ++im) {
            const char* p = sA + (size_t)(arow0 + im * 16) * ROWPITCH + half * 16;
            a[im].q[0] = *(const uint4*)p;
            a[im].q[1] = *(const uint4*)(p + 32);
        }
        for (int in = 0; in < 2; ++in) {
            const char* p = sB + (size_t)(brow0 + in * 16) * ROWPITCH + half * 16;
            bb[in].q[0] = *(const uint4*)p;
            bb[in].q[1] = *(const uint4*)(p + 32);
        }
        for (int im = 0; im < 2; ++im)
            for (int in = 0; in < 2; ++in)
                acc[im][in] = __builtin_amdgcn_wmma_f32_16x16x32_bf16(
                    false, a[im].v, false, bb[in].v,
                    (short)0, acc[im][in], false, false);

        __syncthreads();   // all waves done reading buffer b before it is re-filled
    }

    // store: lane holds column (n_tile + r), rows m_tile + half*8 + e
    for (int im = 0; im < 2; ++im)
        for (int in = 0; in < 2; ++in) {
            int col = n_base + in * 16 + r;
            int row0 = m_base + im * 16 + half * 8;
            for (int e = 0; e < 8; ++e)
                xg[(size_t)(row0 + e) * G4 + col] = acc[im][in][e];
        }
}

// ---------------- LSTM recurrence (one persistent WG per direction) ----------------
__global__ __launch_bounds__(1024)
void lstm_kernel(const float* __restrict__ xg_f, const float* __restrict__ xg_b,
                 const __bf16* __restrict__ whh_f, const __bf16* __restrict__ whh_b,
                 const float* __restrict__ h0, const float* __restrict__ c0,
                 float* __restrict__ hs_f, float* __restrict__ hs_b) {
    __shared__ float sh[Hh];
    __shared__ float sc[Hh];
    __shared__ float sg[G4];

    const int dir = blockIdx.x;
    const float*    xg = dir ? xg_b : xg_f;
    const unsigned* W  = (const unsigned*)(dir ? whh_b : whh_f);  // bf16 pairs
    float*          hs = dir ? hs_b : hs_f;

    const int tid = threadIdx.x;
    if (tid < Hh) { sh[tid] = h0[dir * Hh + tid]; sc[tid] = c0[dir * Hh + tid]; }
    __syncthreads();

    const int row0 = tid;             // gate rows owned by this thread
    const int row1 = tid + 1024;
    const unsigned* w0 = W + (size_t)row0 * (Hh / 2);
    const unsigned* w1 = W + (size_t)row1 * (Hh / 2);

    for (int s = 0; s < SEQ; ++s) {
        const int t = dir ? (SEQ - 1 - s) : s;
        float a0 = 0.f, a1 = 0.f;
        for (int k = 0; k < Hh / 2; ++k) {
            unsigned p0 = w0[k], p1 = w1[k];
            float hk0 = sh[2 * k], hk1 = sh[2 * k + 1];
            a0 = fmaf(__uint_as_float(p0 << 16),          hk0, a0);
            a0 = fmaf(__uint_as_float(p0 & 0xffff0000u),  hk1, a0);
            a1 = fmaf(__uint_as_float(p1 << 16),          hk0, a1);
            a1 = fmaf(__uint_as_float(p1 & 0xffff0000u),  hk1, a1);
        }
        const float* xgr = xg + (size_t)t * G4;
        sg[row0] = a0 + xgr[row0];
        sg[row1] = a1 + xgr[row1];
        __syncthreads();
        if (tid < Hh) {
            float gi = sg[tid], gf = sg[Hh + tid];
            float gg = sg[2 * Hh + tid], go = sg[3 * Hh + tid];
            float i = 1.f / (1.f + __expf(-gi));
            float f = 1.f / (1.f + __expf(-gf));
            float g = tanhf(gg);
            float o = 1.f / (1.f + __expf(-go));
            float c = f * sc[tid] + i * g;
            float h = o * tanhf(c);
            sc[tid] = c;
            sh[tid] = h;
            hs[(size_t)t * Hh + tid] = h;
        }
        __syncthreads();
    }
}

// ---------------- feats = concat(hs_f, hs_b) @ W_out^T + b_out ----------------
__global__ void feats_kernel(const float* __restrict__ hs_f, const float* __restrict__ hs_b,
                             const float* __restrict__ Wout, const float* __restrict__ bout,
                             float* __restrict__ feats) {
    int t = blockIdx.x;
    int j = threadIdx.x;
    if (j >= NTAG) return;
    const float* wr = Wout + (size_t)j * HID;
    const float* hf = hs_f + (size_t)t * Hh;
    const float* hb = hs_b + (size_t)t * Hh;
    float a = bout[j];
    for (int k = 0; k < Hh; ++k) a = fmaf(wr[k],      hf[k], a);
    for (int k = 0; k < Hh; ++k) a = fmaf(wr[Hh + k], hb[k], a);
    feats[(size_t)t * NTAG + j] = a;
}

// ---------------- Viterbi forward + backtrace ----------------
__global__ __launch_bounds__(64)
void viterbi_kernel(const float* __restrict__ feats, const float* __restrict__ logT,
                    int* __restrict__ bp, float* __restrict__ out) {
    __shared__ float T[NTAG * NTAG];
    __shared__ float fv[NTAG];
    __shared__ float fvn[NTAG];
    const int tid = threadIdx.x;
    for (int i = tid; i < NTAG * NTAG; i += blockDim.x) T[i] = logT[i];
    if (tid < NTAG) fv[tid] = (tid == START) ? 0.f : NEGV;
    __syncthreads();

    for (int t = 0; t < SEQ; ++t) {
        if (tid < NTAG) {
            float m = -3.0e38f;
            int arg = 0;
            for (int i = 0; i < NTAG; ++i) {
                float s = fv[i] + T[tid * NTAG + i];
                if (s > m) { m = s; arg = i; }
            }
            fvn[tid] = m + feats[(size_t)t * NTAG + tid];
            bp[(size_t)t * NTAG + tid] = arg;
        }
        __syncthreads();
        if (tid < NTAG) fv[tid] = fvn[tid];
        __syncthreads();
    }

    if (tid == 0) {
        float best = -3.0e38f;
        int bt = 0;
        for (int j = 0; j < NTAG; ++j) {
            float s = fv[j] + T[STOP * NTAG + j];
            if (s > best) { best = s; bt = j; }
        }
        out[0] = best;                 // path_score
        int tag = bt;
        out[1 + SEQ - 1] = (float)tag; // best_path[SEQ-1]
        for (int t = SEQ - 1; t >= 1; --t) {
            tag = bp[(size_t)t * NTAG + tag];
            out[t] = (float)tag;       // out[1 + (t-1)]
        }
    }
}

// ---------------- driver ----------------
extern "C" void kernel_launch(void* const* d_in, const int* in_sizes, int n_in,
                              void* d_out, int out_size, void* d_ws, size_t ws_size,
                              hipStream_t stream) {
    const int*   sentence = (const int*)  d_in[0];
    const float* embed    = (const float*)d_in[1];
    const float* w_ih_f   = (const float*)d_in[2];
    const float* w_hh_f   = (const float*)d_in[3];
    const float* b_ih_f   = (const float*)d_in[4];
    const float* b_hh_f   = (const float*)d_in[5];
    const float* w_ih_b   = (const float*)d_in[6];
    const float* w_hh_b   = (const float*)d_in[7];
    const float* b_ih_b   = (const float*)d_in[8];
    const float* b_hh_b   = (const float*)d_in[9];
    const float* W_out    = (const float*)d_in[10];
    const float* b_out    = (const float*)d_in[11];
    const float* logT     = (const float*)d_in[12];
    const float* h0       = (const float*)d_in[13];
    const float* c0       = (const float*)d_in[14];

    // workspace carve-out
    char* ws = (char*)d_ws;
    size_t off = 0;
    auto carve = [&](size_t bytes) -> void* {
        void* p = ws + off;
        off += (bytes + 255) & ~(size_t)255;
        return p;
    };
    __bf16* x_bf    = (__bf16*)carve((size_t)SEQ * EMB * 2);   // 8 MB
    __bf16* wihf_bf = (__bf16*)carve((size_t)G4 * EMB * 2);    // 4 MB
    __bf16* wihb_bf = (__bf16*)carve((size_t)G4 * EMB * 2);    // 4 MB
    __bf16* whhf_bf = (__bf16*)carve((size_t)G4 * Hh * 2);     // 2 MB
    __bf16* whhb_bf = (__bf16*)carve((size_t)G4 * Hh * 2);     // 2 MB
    float*  xg_f    = (float*) carve((size_t)SEQ * G4 * 4);    // 32 MB
    float*  xg_b    = (float*) carve((size_t)SEQ * G4 * 4);    // 32 MB
    float*  hs_f    = (float*) carve((size_t)SEQ * Hh * 4);    // 8 MB
    float*  hs_b    = (float*) carve((size_t)SEQ * Hh * 4);    // 8 MB
    float*  feats   = (float*) carve((size_t)SEQ * NTAG * 4);
    int*    bp      = (int*)   carve((size_t)SEQ * NTAG * 4);

    // 1) weights -> bf16
    int n;
    n = G4 * EMB;
    f32_to_bf16_kernel<<<(n + 255) / 256, 256, 0, stream>>>(w_ih_f, wihf_bf, n);
    f32_to_bf16_kernel<<<(n + 255) / 256, 256, 0, stream>>>(w_ih_b, wihb_bf, n);
    n = G4 * Hh;
    f32_to_bf16_kernel<<<(n + 255) / 256, 256, 0, stream>>>(w_hh_f, whhf_bf, n);
    f32_to_bf16_kernel<<<(n + 255) / 256, 256, 0, stream>>>(w_hh_b, whhb_bf, n);

    // 2) embedding gather
    embed_gather_kernel<<<SEQ, 256, 0, stream>>>(sentence, embed, x_bf);

    // 3) both gate GEMMs via TDM + WMMA
    dim3 gg(SEQ / 64, G4 / 128, 2);
    gate_gemm_kernel<<<gg, 256, 0, stream>>>(x_bf, wihf_bf, wihb_bf,
                                             b_ih_f, b_hh_f, b_ih_b, b_hh_b,
                                             xg_f, xg_b);

    // 4) bidirectional LSTM scan (block 0 = forward, block 1 = backward)
    lstm_kernel<<<2, 1024, 0, stream>>>(xg_f, xg_b, whhf_bf, whhb_bf,
                                        h0, c0, hs_f, hs_b);

    // 5) output projection
    feats_kernel<<<SEQ, 64, 0, stream>>>(hs_f, hs_b, W_out, b_out, feats);

    // 6) Viterbi + backtrace
    viterbi_kernel<<<1, 64, 0, stream>>>(feats, logT, bp, (float*)d_out);
}
